// GNN22_46093589020764
// MI455X (gfx1250) — matrix operand order repaired and checked by reading, not compiled, not run on hardware
//
#include <hip/hip_runtime.h>
#include <hip/hip_bf16.h>
#include <math.h>

typedef __attribute__((ext_vector_type(16))) _Float16 v16h;
typedef __attribute__((ext_vector_type(8)))  _Float16 v8h;
typedef __attribute__((ext_vector_type(8)))  float    v8f;

#define ACT_RELU    0
#define ACT_LEAKY   1
#define ACT_SIGMOID 2
#define NEG_SLOPE   0.01f

// ---------------------------------------------------------------------------
// f16 fragment loaders (operands pre-converted to f16 in workspace).
//
// A matrix 16x32 (MxK), k order used here (contiguous, not the ISA table order; B uses the same):
//   lanes 0-15 : row M=lane,    halves = K[0..15]
//   lanes 16-31: row M=lane-16, halves = K[16..31]
// => lane reads 16 contiguous halves at k0 + half_sel*16 (2 x 16B).
__device__ __forceinline__ v16h load_a_frag(const _Float16* __restrict__ A, int K,
                                            int row, int k0, int half_sel) {
  const v8h* p = (const v8h*)(A + (size_t)row * K + k0 + half_sel * 16);
  v8h lo = p[0];
  v8h hi = p[1];              // +8 halves
  v16h f;
#pragma unroll
  for (int i = 0; i < 8; ++i) { f[i] = lo[i]; f[8 + i] = hi[i]; }
  return f;
}

// B matrix 32x16 (KxN) with B[k][n] = W[n][k] (W is [N,K] row-major):
//   lanes 0-15 : col N=lane,    halves = K[0..15]
//   lanes 16-31: col N=lane-16, halves = K[16..31]
// => lane reads 16 contiguous halves (32B) of W row `ncol` at k0 + half_sel*16.
__device__ __forceinline__ v16h load_w_frag(const _Float16* __restrict__ W, int K,
                                            int N, int ncol, int k0, int half_sel) {
  if (ncol < N) {
    return *(const v16h*)(W + (size_t)ncol * K + k0 + half_sel * 16);
  }
  v16h z;
#pragma unroll
  for (int i = 0; i < 16; ++i) z[i] = (_Float16)0.0f;
  return z;
}

__device__ __forceinline__ void store_tile(v8f acc, const float* __restrict__ bias,
                                           int rbase, int col, int M, int N,
                                           int act, int out_half, void* __restrict__ out) {
  if (col >= N) return;                 // per-lane; all WMMAs already done
  const float bv = bias[col];
#pragma unroll
  for (int r = 0; r < 8; ++r) {
    const int rr = rbase + r;
    if (rr >= M) continue;
    float v = acc[r] + bv;
    if (act == ACT_RELU)       v = v > 0.0f ? v : 0.0f;
    else if (act == ACT_LEAKY) v = v > 0.0f ? v : NEG_SLOPE * v;
    else                       v = 1.0f / (1.0f + __expf(-v));
    if (out_half) ((_Float16*)out)[(size_t)rr * N + col] = (_Float16)v;
    else          ((float*)out)   [(size_t)rr * N + col] = v;
  }
}

// ---------------------------------------------------------------------------
// out[M,N] = act( A[M,K] @ WA^T  (+ B[M,K] @ WB^T)  + bias )
// One wave computes a 32x32 output tile: 2 A-frags x 2 W-frags -> 4
// accumulators, 4 WMMAs per 32-wide K step. f16 operands, f32 accumulate.
__global__ void __launch_bounds__(32)
wmma_gemm_kernel(const _Float16* __restrict__ A,  const _Float16* __restrict__ WA,
                 const _Float16* __restrict__ B,  const _Float16* __restrict__ WB,
                 const float* __restrict__ bias, void* __restrict__ out,
                 int M, int N, int K, int act, int out_half) {
  const int lane     = threadIdx.x & 31;
  const int half_sel = lane >> 4;
  const int l        = lane & 15;

  const int m0 = blockIdx.x * 32;
  const int n0 = blockIdx.y * 32;

  int row0 = m0 + l;       if (row0 >= M) row0 = M - 1;   // clamp, stores guarded
  int row1 = m0 + 16 + l;  if (row1 >= M) row1 = M - 1;
  const int col0 = n0 + l;
  const int col1 = n0 + 16 + l;

  v8f c00 = {}, c01 = {}, c10 = {}, c11 = {};
  for (int k0 = 0; k0 < K; k0 += 32) {
    v16h a0 = load_a_frag(A, K, row0, k0, half_sel);
    v16h a1 = load_a_frag(A, K, row1, k0, half_sel);
    v16h w0 = load_w_frag(WA, K, N, col0, k0, half_sel);
    v16h w1 = load_w_frag(WA, K, N, col1, k0, half_sel);
    c00 = __builtin_amdgcn_wmma_f32_16x16x32_f16(false, a0, false, w0, (short)0, c00, false, false);
    c01 = __builtin_amdgcn_wmma_f32_16x16x32_f16(false, a0, false, w1, (short)0, c01, false, false);
    c10 = __builtin_amdgcn_wmma_f32_16x16x32_f16(false, a1, false, w0, (short)0, c10, false, false);
    c11 = __builtin_amdgcn_wmma_f32_16x16x32_f16(false, a1, false, w1, (short)0, c11, false, false);
  }
  if (B != nullptr) {
    for (int k0 = 0; k0 < K; k0 += 32) {
      v16h a0 = load_a_frag(B, K, row0, k0, half_sel);
      v16h a1 = load_a_frag(B, K, row1, k0, half_sel);
      v16h w0 = load_w_frag(WB, K, N, col0, k0, half_sel);
      v16h w1 = load_w_frag(WB, K, N, col1, k0, half_sel);
      c00 = __builtin_amdgcn_wmma_f32_16x16x32_f16(false, a0, false, w0, (short)0, c00, false, false);
      c01 = __builtin_amdgcn_wmma_f32_16x16x32_f16(false, a0, false, w1, (short)0, c01, false, false);
      c10 = __builtin_amdgcn_wmma_f32_16x16x32_f16(false, a1, false, w0, (short)0, c10, false, false);
      c11 = __builtin_amdgcn_wmma_f32_16x16x32_f16(false, a1, false, w1, (short)0, c11, false, false);
    }
  }

  // C/D layout: VGPR r -> (lanes 0-15: M=r, lanes 16-31: M=8+r), N = lane%16.
  const int rb0 = m0 + half_sel * 8;
  const int rb1 = m0 + 16 + half_sel * 8;
  store_tile(c00, bias, rb0, col0, M, N, act, out_half, out);
  store_tile(c01, bias, rb0, col1, M, N, act, out_half, out);
  store_tile(c10, bias, rb1, col0, M, N, act, out_half, out);
  store_tile(c11, bias, rb1, col1, M, N, act, out_half, out);
}

// ---------------------------------------------------------------------------
// Segment-max over edges. pooled values are post-relu (>= 0): for non-negative
// IEEE floats u32 ordering == float ordering, so one global_atomic_max_u32
// per edge-feature. Accumulator pre-zeroed, which also realizes the
// reference's isneginf -> 0 rule for isolated nodes.
__global__ void edge_max_kernel(const float* __restrict__ pooled,
                                const int* __restrict__ src,
                                const int* __restrict__ dst,
                                unsigned int* __restrict__ agg,
                                int E, int F) {
  const int f = threadIdx.x;                          // feature lane, coalesced
  const int e = blockIdx.x * blockDim.y + threadIdx.y;
  if (e >= E) return;
  const int s = src[e];
  const int d = dst[e];
  const float v = pooled[(size_t)s * F + f];
  atomicMax(agg + (size_t)d * F + f, __float_as_uint(v));
}

__global__ void zero_kernel(float* __restrict__ p, size_t n) {
  size_t i = (size_t)blockIdx.x * blockDim.x + threadIdx.x;
  if (i < n) p[i] = 0.0f;
}

__global__ void cvt_f32_f16_kernel(const float* __restrict__ in,
                                   _Float16* __restrict__ out, size_t n) {
  size_t i = (size_t)blockIdx.x * blockDim.x + threadIdx.x;
  if (i < n) out[i] = (_Float16)in[i];
}

// ---------------------------------------------------------------------------
extern "C" void kernel_launch(void* const* d_in, const int* in_sizes, int n_in,
                              void* d_out, int out_size, void* d_ws, size_t ws_size,
                              hipStream_t stream) {
  const float* feat    = (const float*)d_in[0];
  const int*   src     = (const int*)  d_in[1];
  const int*   dst     = (const int*)  d_in[2];
  const float* Wp1     = (const float*)d_in[3];
  const float* bp1     = (const float*)d_in[4];
  const float* Wself1  = (const float*)d_in[5];
  const float* Wneigh1 = (const float*)d_in[6];
  const float* bneigh1 = (const float*)d_in[7];
  const float* Wp2     = (const float*)d_in[8];
  const float* bp2     = (const float*)d_in[9];
  const float* Wself2  = (const float*)d_in[10];
  const float* Wneigh2 = (const float*)d_in[11];
  const float* bneigh2 = (const float*)d_in[12];
  const float* W1      = (const float*)d_in[13];
  const float* b1      = (const float*)d_in[14];
  const float* W2      = (const float*)d_in[15];
  const float* b2      = (const float*)d_in[16];

  const int IN = 64, H1 = 128, H2 = 128, H3 = 256, C = 100;
  const int Nn = in_sizes[0] / IN;   // 50000
  const int E  = in_sizes[1];        // 1600000

  // ---- workspace layout ----
  // f32 region: pooled [Nn*128], hneighF [Nn*128]
  // f16 region: featH, h1H, h2H, h3H, hneighH, 8 weight matrices
  float* wsf = (float*)d_ws;
  const size_t NBf = (size_t)Nn * 128;
  float* pooled  = wsf;
  float* hneighF = wsf + NBf;
  _Float16* p16 = (_Float16*)(wsf + 2 * NBf);
  _Float16* featH   = p16; p16 += (size_t)Nn * IN;
  _Float16* h1H     = p16; p16 += (size_t)Nn * H1;
  _Float16* h2H     = p16; p16 += (size_t)Nn * H2;
  _Float16* h3H     = p16; p16 += (size_t)Nn * H3;
  _Float16* hneighH = p16; p16 += (size_t)Nn * H1;
  _Float16* Wp1H    = p16; p16 += (size_t)IN * IN;
  _Float16* Wself1H = p16; p16 += (size_t)H1 * IN;
  _Float16* Wneigh1H= p16; p16 += (size_t)H1 * IN;
  _Float16* Wp2H    = p16; p16 += (size_t)H1 * H1;
  _Float16* Wself2H = p16; p16 += (size_t)H2 * H1;
  _Float16* Wneigh2H= p16; p16 += (size_t)H2 * H1;
  _Float16* W1H     = p16; p16 += (size_t)H3 * H2;
  _Float16* W2H     = p16; p16 += (size_t)C * H3;

  auto cvt = [&](const float* s, _Float16* d, size_t n) {
    cvt_f32_f16_kernel<<<(unsigned)((n + 255) / 256), 256, 0, stream>>>(s, d, n);
  };
  auto zero = [&](float* p, size_t n) {
    zero_kernel<<<(unsigned)((n + 255) / 256), 256, 0, stream>>>(p, n);
  };
  auto gemm_grid = [&](int N_) { return dim3((Nn + 31) / 32, (N_ + 31) / 32); };
  const dim3 blk(32);

  // ---- one-time f16 conversions ----
  cvt(feat, featH, (size_t)Nn * IN);
  cvt(Wp1, Wp1H, (size_t)IN * IN);
  cvt(Wself1, Wself1H, (size_t)H1 * IN);
  cvt(Wneigh1, Wneigh1H, (size_t)H1 * IN);
  cvt(Wp2, Wp2H, (size_t)H1 * H1);
  cvt(Wself2, Wself2H, (size_t)H2 * H1);
  cvt(Wneigh2, Wneigh2H, (size_t)H2 * H1);
  cvt(W1, W1H, (size_t)H3 * H2);
  cvt(W2, W2H, (size_t)C * H3);

  // ---- layer 1 ----
  wmma_gemm_kernel<<<gemm_grid(IN), blk, 0, stream>>>(
      featH, Wp1H, nullptr, nullptr, bp1, pooled, Nn, IN, IN, ACT_RELU, 0);
  zero(hneighF, (size_t)Nn * IN);
  {
    dim3 b(IN, 256 / IN);
    edge_max_kernel<<<(E + (int)b.y - 1) / (int)b.y, b, 0, stream>>>(
        pooled, src, dst, (unsigned int*)hneighF, E, IN);
  }
  cvt(hneighF, hneighH, (size_t)Nn * IN);
  wmma_gemm_kernel<<<gemm_grid(H1), blk, 0, stream>>>(
      featH, Wself1H, hneighH, Wneigh1H, bneigh1, h1H, Nn, H1, IN, ACT_LEAKY, 1);

  // ---- layer 2 ----
  wmma_gemm_kernel<<<gemm_grid(H1), blk, 0, stream>>>(
      h1H, Wp2H, nullptr, nullptr, bp2, pooled, Nn, H1, H1, ACT_RELU, 0);
  zero(hneighF, (size_t)Nn * H1);
  {
    dim3 b(H1, 256 / H1);
    edge_max_kernel<<<(E + (int)b.y - 1) / (int)b.y, b, 0, stream>>>(
        pooled, src, dst, (unsigned int*)hneighF, E, H1);
  }
  cvt(hneighF, hneighH, (size_t)Nn * H1);
  wmma_gemm_kernel<<<gemm_grid(H2), blk, 0, stream>>>(
      h1H, Wself2H, hneighH, Wneigh2H, bneigh2, h2H, Nn, H2, H1, ACT_LEAKY, 1);

  // ---- MLP head ----
  wmma_gemm_kernel<<<gemm_grid(H3), blk, 0, stream>>>(
      h2H, W1H, nullptr, nullptr, b1, h3H, Nn, H3, H2, ACT_LEAKY, 1);
  wmma_gemm_kernel<<<gemm_grid(C), blk, 0, stream>>>(
      h3H, W2H, nullptr, nullptr, b2, d_out, Nn, C, H3, ACT_SIGMOID, 0);
}
